// DetectionLayer_21895743275509
// MI455X (gfx1250) — compile-verified, matrix operand order checked by reading, not run on hardware
//
#include <hip/hip_runtime.h>
#include <hip/hip_bf16.h>

#define N_ROIS      5000
#define NUM_CLASSES 81
#define MIN_CONF    0.7f
#define MAX_INST    100
#define NMS_THR     0.3f

typedef __attribute__((ext_vector_type(16))) _Float16 v16h;
typedef __attribute__((ext_vector_type(8)))  float    v8f;

// ---------------------------------------------------------------------------
// Kernel 1: per-ROI argmax over 81 classes, delta refine, clip, validity key.
// ---------------------------------------------------------------------------
__global__ void refine_kernel(const float* __restrict__ rois,
                              const float* __restrict__ probs,
                              const float* __restrict__ deltas,
                              const float* __restrict__ window,
                              float4* __restrict__ refined,
                              float*  __restrict__ scores,
                              int*    __restrict__ cls,
                              float*  __restrict__ keys) {
  int i = blockIdx.x * blockDim.x + threadIdx.x;
  if (i >= N_ROIS) return;

  const float* p = probs + (size_t)i * NUM_CLASSES;
  int   best_c = 0;
  float best   = p[0];
  for (int c = 1; c < NUM_CLASSES; ++c) {       // first-max tie-break == argmax
    float v = p[c];
    if (v > best) { best = v; best_c = c; }
  }

  const float* d = deltas + ((size_t)i * NUM_CLASSES + best_c) * 4;
  float dy = d[0] * 0.1f, dx = d[1] * 0.1f;
  float dh = d[2] * 0.2f, dw = d[3] * 0.2f;

  float y1 = rois[i * 4 + 0], x1 = rois[i * 4 + 1];
  float y2 = rois[i * 4 + 2], x2 = rois[i * 4 + 3];
  float h = y2 - y1, w = x2 - x1;
  float cy = y1 + 0.5f * h + dy * h;
  float cx = x1 + 0.5f * w + dx * w;
  h *= expf(dh);
  w *= expf(dw);
  float ny1 = cy - 0.5f * h, nx1 = cx - 0.5f * w;
  float ny2 = ny1 + h,       nx2 = nx1 + w;

  float wy1 = window[0], wx1 = window[1], wy2 = window[2], wx2 = window[3];
  ny1 = fminf(fmaxf(ny1, wy1), wy2);
  nx1 = fminf(fmaxf(nx1, wx1), wx2);
  ny2 = fminf(fmaxf(ny2, wy1), wy2);
  nx2 = fminf(fmaxf(nx2, wx1), wx2);

  refined[i] = make_float4(ny1, nx1, ny2, nx2);
  scores[i]  = best;
  cls[i]     = best_c;
  bool valid = (best_c > 0) && (best >= MIN_CONF);
  keys[i]    = valid ? best : -INFINITY;
}

// ---------------------------------------------------------------------------
// Kernel 2: stable descending rank via WMMA column-sum of indicator matrix.
// rank[i] = #{j: key_j > key_i} + #{j < i: key_j == key_i}; order[rank] = i.
// One wave per 16 columns. A = ones(16x32 f16); B[k][n] = indicator. Since A
// is all ones, D[0][n] = sum_k B[k][n] regardless of K packing — only the
// column<->lane mapping matters (lane n and lane n+16 both feed column n).
// Branchless: full chunks need no bounds checks; tail chunk uses value
// selects only, so EXEC stays all-1s through every WMMA.
// ---------------------------------------------------------------------------
__global__ void __launch_bounds__(32)
rank_kernel(const float* __restrict__ keys, int* __restrict__ order) {
  const int lane  = threadIdx.x;               // wave32: 0..31
  const int col   = lane & 15;
  const int i_col = blockIdx.x * 16 + col;
  const int i_cl  = (i_col < N_ROIS) ? i_col : (N_ROIS - 1);
  const float key_i = keys[i_cl];

  const _Float16 one  = (_Float16)1.0f;
  const _Float16 zero = (_Float16)0.0f;

  v16h a;
#pragma unroll
  for (int t = 0; t < 16; ++t) a[t] = one;

  v8f acc = {0.f, 0.f, 0.f, 0.f, 0.f, 0.f, 0.f, 0.f};

  const int jHalf = lane & 16;                 // lanes 16..31: second 16 j's
  const float4* __restrict__ keys4 = (const float4*)keys;

  // ---- full chunks: j in [0, 4992), no bounds checks ---------------------
  const int FULL = (N_ROIS / 32) * 32;         // 4992
  for (int jBase = 0; jBase < FULL; jBase += 32) {
    const int j0 = jBase + jHalf;              // multiple of 16 floats (64 B)
    const int q  = j0 >> 2;
    float4 k0 = keys4[q + 0];
    float4 k1 = keys4[q + 1];
    float4 k2 = keys4[q + 2];
    float4 k3 = keys4[q + 3];
    float kv[16] = {k0.x, k0.y, k0.z, k0.w, k1.x, k1.y, k1.z, k1.w,
                    k2.x, k2.y, k2.z, k2.w, k3.x, k3.y, k3.z, k3.w};
    v16h b;
#pragma unroll
    for (int t = 0; t < 16; ++t) {
      const int  j   = j0 + t;
      const bool gt  = kv[t] > key_i;
      const bool tie = (kv[t] == key_i) & (j < i_col);
      b[t] = (gt | tie) ? one : zero;
    }
    acc = __builtin_amdgcn_wmma_f32_16x16x32_f16(
        false, a, false, b, (short)0, acc, false, false);
  }

  // ---- tail chunk: j in [4992, 5024), value-select clamping only ---------
  {
    const int j0 = FULL + jHalf;
    v16h b;
#pragma unroll
    for (int t = 0; t < 16; ++t) {
      const int  j   = j0 + t;
      const int  jc  = (j < N_ROIS) ? j : 0;   // value select, not a branch
      const float kj = keys[jc];
      const bool inb = (j < N_ROIS);
      const bool gt  = kj > key_i;
      const bool tie = (kj == key_i) & (j < i_col);
      b[t] = (inb & (gt | tie)) ? one : zero;
    }
    acc = __builtin_amdgcn_wmma_f32_16x16x32_f16(
        false, a, false, b, (short)0, acc, false, false);
  }

  // D layout: VGPR0, lanes 0..15 hold M=0, N=lane -> column sums.
  if (lane < 16 && i_col < N_ROIS) {
    int r = (int)(acc[0] + 0.5f);              // exact integer in f32
    order[r] = i_col;
  }
}

// ---------------------------------------------------------------------------
// Kernel 3: single-block sequential greedy NMS (exact scan semantics) + emit.
// Boxes staged global->LDS with CDNA5 async loads. Kept list is compacted in
// processing order == score-desc/index-asc == lax.top_k order.
// ---------------------------------------------------------------------------
__global__ void __launch_bounds__(1024)
nms_kernel(const float4* __restrict__ g_refined,
           const float*  __restrict__ g_scores,
           const int*    __restrict__ g_cls,
           const int*    __restrict__ g_order,
           float*        __restrict__ out) {
  extern __shared__ unsigned char smem[];
  float4* boxes      = (float4*)smem;                              // 80000 B
  int*    keptPacked = (int*)(smem + N_ROIS * sizeof(float4));     // 20000 B

  __shared__ int counts[NUM_CLASSES];
  __shared__ int sh_i, sh_c, sh_sup, sh_keptCount;

  const int tid  = threadIdx.x;
  const int bdim = blockDim.x;

  // --- async stage refined boxes into LDS (ASYNCcnt path) -----------------
  for (int idx = tid; idx < N_ROIS; idx += bdim) {
    unsigned int lds_off =
        (unsigned int)(unsigned long long)(uintptr_t)(&boxes[idx]);
    unsigned long long ga = (unsigned long long)(uintptr_t)(&g_refined[idx]);
    asm volatile("global_load_async_to_lds_b128 %0, %1, off"
                 :: "v"(lds_off), "v"(ga) : "memory");
  }
  asm volatile("s_wait_asynccnt 0" ::: "memory");   // each wave drains its own
  if (tid < NUM_CLASSES) counts[tid] = 0;
  if (tid == 0) { sh_keptCount = 0; sh_sup = 0; }
  __syncthreads();                                   // all waves' loads done

  // --- sequential scan, two barriers per step ------------------------------
  int  pend_i = -1, pend_c = 0;
  bool pend_valid = false;

  for (int idx = 0; idx < N_ROIS; ++idx) {
    if (tid == 0) {
      // commit previous candidate using last round's suppression verdict
      if (pend_i >= 0) {
        bool ok = pend_valid && (sh_sup == 0) && (counts[pend_c] < MAX_INST);
        if (ok) {
          keptPacked[sh_keptCount] = pend_i | (pend_c << 16);
          sh_keptCount++;
          counts[pend_c]++;
        }
      }
      int   i = g_order[idx];
      int   c = g_cls[i];
      float s = g_scores[i];
      pend_i = i; pend_c = c;
      pend_valid = (c > 0) && (s >= MIN_CONF);
      sh_i = i; sh_c = c; sh_sup = 0;
    }
    __syncthreads();

    const int i  = sh_i;
    const int c  = sh_c;
    const int kc = sh_keptCount;
    const float4 bi = boxes[i];
    const float area_i =
        fmaxf(bi.z - bi.x, 0.f) * fmaxf(bi.w - bi.y, 0.f);

    bool sup = false;
    for (int t = tid; t < kc; t += bdim) {
      int p = keptPacked[t];
      if ((p >> 16) == c) {
        float4 bk = boxes[p & 0xFFFF];
        float area_k = fmaxf(bk.z - bk.x, 0.f) * fmaxf(bk.w - bk.y, 0.f);
        float iy1 = fmaxf(bi.x, bk.x), ix1 = fmaxf(bi.y, bk.y);
        float iy2 = fminf(bi.z, bk.z), ix2 = fminf(bi.w, bk.w);
        float inter = fmaxf(iy2 - iy1, 0.f) * fmaxf(ix2 - ix1, 0.f);
        float uni   = area_i + area_k - inter;
        float iou   = (uni > 0.f) ? (inter / fmaxf(uni, 1e-12f)) : 0.f;
        if (iou > NMS_THR) { sup = true; break; }
      }
    }
    if (sup) atomicOr(&sh_sup, 1);
    __syncthreads();
  }

  if (tid == 0) {                                    // commit final candidate
    if (pend_i >= 0) {
      bool ok = pend_valid && (sh_sup == 0) && (counts[pend_c] < MAX_INST);
      if (ok) {
        keptPacked[sh_keptCount] = pend_i | (pend_c << 16);
        sh_keptCount++;
        counts[pend_c]++;
      }
    }
  }
  __syncthreads();

  // --- emit top-100 rows: first kept entries, zeros otherwise --------------
  if (tid < MAX_INST) {
    float o0 = 0.f, o1 = 0.f, o2 = 0.f, o3 = 0.f, o4 = 0.f, o5 = 0.f;
    if (tid < sh_keptCount) {
      int p = keptPacked[tid];
      int i = p & 0xFFFF;
      int c = p >> 16;
      float4 b = boxes[i];
      o0 = b.x; o1 = b.y; o2 = b.z; o3 = b.w;
      o4 = (float)c; o5 = g_scores[i];
    }
    out[tid * 6 + 0] = o0; out[tid * 6 + 1] = o1; out[tid * 6 + 2] = o2;
    out[tid * 6 + 3] = o3; out[tid * 6 + 4] = o4; out[tid * 6 + 5] = o5;
  }
}

// ---------------------------------------------------------------------------
extern "C" void kernel_launch(void* const* d_in, const int* in_sizes, int n_in,
                              void* d_out, int out_size, void* d_ws,
                              size_t ws_size, hipStream_t stream) {
  const float* rois   = (const float*)d_in[0];   // [5000,4]
  const float* probs  = (const float*)d_in[1];   // [5000,81]
  const float* deltas = (const float*)d_in[2];   // [5000,81,4]
  const float* window = (const float*)d_in[3];   // [4]

  unsigned char* ws = (unsigned char*)d_ws;
  float4* refined = (float4*)(ws + 0);           // 80000 B
  float*  scores  = (float*) (ws + 80000);       // 20000 B
  int*    cls     = (int*)   (ws + 100000);      // 20000 B
  float*  keys    = (float*) (ws + 120000);      // 20000 B (16B aligned)
  int*    order   = (int*)   (ws + 140000);      // 20000 B
  float*  out     = (float*)d_out;               // 100*6

  refine_kernel<<<(N_ROIS + 255) / 256, 256, 0, stream>>>(
      rois, probs, deltas, window, refined, scores, cls, keys);

  rank_kernel<<<(N_ROIS + 15) / 16, 32, 0, stream>>>(keys, order);

  const size_t lds_bytes = N_ROIS * sizeof(float4) + N_ROIS * sizeof(int);
  (void)hipFuncSetAttribute((const void*)nms_kernel,
                            hipFuncAttributeMaxDynamicSharedMemorySize,
                            (int)lds_bytes);     // 320KB/WGP on CDNA5
  nms_kernel<<<1, 1024, lds_bytes, stream>>>(refined, scores, cls, order, out);
}
